// CNNMoE_29111288332647
// MI455X (gfx1250) — matrix operand order repaired and checked by reading, not compiled, not run on hardware
//
#include <hip/hip_runtime.h>

typedef __attribute__((ext_vector_type(16))) _Float16 v16h;
typedef __attribute__((ext_vector_type(8)))  _Float16 v8h;
typedef __attribute__((ext_vector_type(2)))  _Float16 v2h;
typedef __attribute__((ext_vector_type(8)))  float    v8f;

static inline int ceilDiv(long long n, int d) { return (int)((n + d - 1) / d); }

// ---------------------------------------------------------------------------
// WMMA GEMM:  C[M,N] = A[M,K] * Bt[N,K]^T  (+bias, += old C, ReLU optional)
// Block tile 64x128 (8 waves), wave tile 32x32 (4 wmma per K-step of 32).
// Software-prefetched global loads, f16 LDS staging, ds_load_b128 fragments.
// Requirements: K % 32 == 0, M % 64 == 0, N % 16 == 0, lda/ldb % 4 == 0.
// ---------------------------------------------------------------------------
#define BM 64
#define BN 128
#define BK 32
#define LDSW 40   // padded row stride in halves (80B: 16B-aligned, skewed banks)

template <int ACCUM, int RELU>
__global__ __launch_bounds__(256)
void gemm_wmma_f16(const float* __restrict__ A, const float* __restrict__ Bt,
                   const float* __restrict__ bias, float* __restrict__ C,
                   int M, int N, int K, int lda, int ldb, int ldc)
{
    __shared__ _Float16 As[BM * LDSW];
    __shared__ _Float16 Bs[BN * LDSW];

    const int tid  = threadIdx.x;
    const int lane = tid & 31;
    const int wave = tid >> 5;
    const int nbn  = (N + BN - 1) / BN;
    const int tm   = (blockIdx.x / nbn) * BM;
    const int tn   = (blockIdx.x % nbn) * BN;

    const int wm   = wave >> 2;          // 0..1  (M sub-block)
    const int wn   = wave & 3;           // 0..3  (N sub-block)
    const int l15  = lane & 15;
    const int half = lane >> 4;

    // staging assignment: A -> 8 floats/thread, B -> 16 floats/thread
    const int ar = tid >> 2;             // 0..63
    const int aq = (tid & 3) << 3;       // 0,8,16,24
    const int br = tid >> 1;             // 0..127
    const int bq = (tid & 1) << 4;       // 0,16

    int gbrow = tn + br; if (gbrow >= N) gbrow = N - 1;   // N may be < BN
    const float* gA = A  + (size_t)(tm + ar) * lda + aq;
    const float* gB = Bt + (size_t)gbrow * ldb + bq;

    v8f acc00 = {}; v8f acc01 = {}; v8f acc10 = {}; v8f acc11 = {};

    // prologue prefetch (k0 = 0)
    float4 fa0 = *(const float4*)(gA);
    float4 fa1 = *(const float4*)(gA + 4);
    float4 fb0 = *(const float4*)(gB);
    float4 fb1 = *(const float4*)(gB + 4);
    float4 fb2 = *(const float4*)(gB + 8);
    float4 fb3 = *(const float4*)(gB + 12);

    for (int k0 = 0; k0 < K; k0 += BK) {
        __syncthreads();                 // previous iteration's LDS reads done
        {
            _Float16* ap = &As[ar * LDSW + aq];
            v2h p;
            p[0] = (_Float16)fa0.x; p[1] = (_Float16)fa0.y; *(v2h*)(ap + 0) = p;
            p[0] = (_Float16)fa0.z; p[1] = (_Float16)fa0.w; *(v2h*)(ap + 2) = p;
            p[0] = (_Float16)fa1.x; p[1] = (_Float16)fa1.y; *(v2h*)(ap + 4) = p;
            p[0] = (_Float16)fa1.z; p[1] = (_Float16)fa1.w; *(v2h*)(ap + 6) = p;
            _Float16* bp = &Bs[br * LDSW + bq];
            p[0] = (_Float16)fb0.x; p[1] = (_Float16)fb0.y; *(v2h*)(bp + 0)  = p;
            p[0] = (_Float16)fb0.z; p[1] = (_Float16)fb0.w; *(v2h*)(bp + 2)  = p;
            p[0] = (_Float16)fb1.x; p[1] = (_Float16)fb1.y; *(v2h*)(bp + 4)  = p;
            p[0] = (_Float16)fb1.z; p[1] = (_Float16)fb1.w; *(v2h*)(bp + 6)  = p;
            p[0] = (_Float16)fb2.x; p[1] = (_Float16)fb2.y; *(v2h*)(bp + 8)  = p;
            p[0] = (_Float16)fb2.z; p[1] = (_Float16)fb2.w; *(v2h*)(bp + 10) = p;
            p[0] = (_Float16)fb3.x; p[1] = (_Float16)fb3.y; *(v2h*)(bp + 12) = p;
            p[0] = (_Float16)fb3.z; p[1] = (_Float16)fb3.w; *(v2h*)(bp + 14) = p;
        }
        __syncthreads();                 // LDS tile ready

        // prefetch next K-step (branchless; final iteration re-fetches k=0)
        int kn = k0 + BK; if (kn >= K) kn = 0;
        float4 na0 = *(const float4*)(gA + kn);
        float4 na1 = *(const float4*)(gA + kn + 4);
        float4 nb0 = *(const float4*)(gB + kn);
        float4 nb1 = *(const float4*)(gB + kn + 4);
        float4 nb2 = *(const float4*)(gB + kn + 8);
        float4 nb3 = *(const float4*)(gB + kn + 12);

        // ---- fragments from LDS + 4 WMMAs ----
        const int ra0 = (wm << 5) + l15;
        const int rb0 = (wn << 5) + l15;
        v8h alo, ahi;
        alo = *(const v8h*)(&As[ra0 * LDSW + (half << 3)]);
        ahi = *(const v8h*)(&As[ra0 * LDSW + 16 + (half << 3)]);
        v16h a0 = __builtin_shufflevector(alo, ahi, 0,1,2,3,4,5,6,7,8,9,10,11,12,13,14,15);
        alo = *(const v8h*)(&As[(ra0 + 16) * LDSW + (half << 3)]);
        ahi = *(const v8h*)(&As[(ra0 + 16) * LDSW + 16 + (half << 3)]);
        v16h a1 = __builtin_shufflevector(alo, ahi, 0,1,2,3,4,5,6,7,8,9,10,11,12,13,14,15);
        v8h blo, bhi;
        blo = *(const v8h*)(&Bs[rb0 * LDSW + (half << 3)]);
        bhi = *(const v8h*)(&Bs[rb0 * LDSW + 16 + (half << 3)]);
        v16h b0 = __builtin_shufflevector(blo, bhi, 0,1,2,3,4,5,6,7,8,9,10,11,12,13,14,15);
        blo = *(const v8h*)(&Bs[(rb0 + 16) * LDSW + (half << 3)]);
        bhi = *(const v8h*)(&Bs[(rb0 + 16) * LDSW + 16 + (half << 3)]);
        v16h b1 = __builtin_shufflevector(blo, bhi, 0,1,2,3,4,5,6,7,8,9,10,11,12,13,14,15);

        acc00 = __builtin_amdgcn_wmma_f32_16x16x32_f16(false, a0, false, b0, (short)0, acc00, false, false);
        acc01 = __builtin_amdgcn_wmma_f32_16x16x32_f16(false, a0, false, b1, (short)0, acc01, false, false);
        acc10 = __builtin_amdgcn_wmma_f32_16x16x32_f16(false, a1, false, b0, (short)0, acc10, false, false);
        acc11 = __builtin_amdgcn_wmma_f32_16x16x32_f16(false, a1, false, b1, (short)0, acc11, false, false);

        fa0 = na0; fa1 = na1;
        fb0 = nb0; fb1 = nb1; fb2 = nb2; fb3 = nb3;
    }

    // ---- epilogue (branch-free inner body; M % 64 == 0 guaranteed) ----
    #pragma unroll
    for (int i = 0; i < 2; ++i) {
        #pragma unroll
        for (int j = 0; j < 2; ++j) {
            v8f accv = (i == 0) ? ((j == 0) ? acc00 : acc01)
                                : ((j == 0) ? acc10 : acc11);
            int cn = tn + (wn << 5) + (j << 4) + l15;
            if (cn < N) {                          // 16-aligned => wave-uniform
                float bv = bias[cn];
                float* cp = C + (size_t)(tm + (wm << 5) + (i << 4) + (half << 3)) * ldc + cn;
                if (ACCUM) {
                    float old[8];
                    #pragma unroll
                    for (int r = 0; r < 8; ++r) old[r] = cp[(size_t)r * ldc];
                    #pragma unroll
                    for (int r = 0; r < 8; ++r) {
                        float val = old[r] + accv[r] + bv;
                        if (RELU) val = fmaxf(val, 0.f);
                        cp[(size_t)r * ldc] = val;
                    }
                } else {
                    #pragma unroll
                    for (int r = 0; r < 8; ++r) {
                        float val = accv[r] + bv;
                        if (RELU) val = fmaxf(val, 0.f);
                        cp[(size_t)r * ldc] = val;
                    }
                }
            }
        }
    }
}

// ---------------------------------------------------------------------------
// Unfold: x[16,3,256,256] -> patches[4096,768]
// ---------------------------------------------------------------------------
__global__ void unfold_kernel(const float* __restrict__ x, float* __restrict__ patches)
{
    long long idx = (long long)blockIdx.x * blockDim.x + threadIdx.x;
    if (idx >= 4096LL * 768) return;
    int d = (int)(idx % 768);
    int t = (int)(idx / 768);
    int c = d >> 8, r = d & 255, iy = r >> 4, ix = r & 15;
    int px = t & 15, py = (t >> 4) & 15, n = t >> 8;
    patches[idx] = x[(((size_t)n * 3 + c) * 256 + py * 16 + iy) * 256 + px * 16 + ix];
}

// pad pw1 weights [64,3] -> [64,32] zero-padded
__global__ void pad_pw1(const float* __restrict__ w, float* __restrict__ out)
{
    int idx = blockIdx.x * blockDim.x + threadIdx.x;
    if (idx >= 2048) return;
    int k = idx & 31, row = idx >> 5;
    out[idx] = (k < 3) ? w[row * 3 + k] : 0.f;
}

// ---------------------------------------------------------------------------
// Gate: top-2 softmax weights into wmat[8][4096] + deterministic l_aux partials
// ---------------------------------------------------------------------------
__global__ __launch_bounds__(256)
void gate_kernel(const float* __restrict__ patches, const float* __restrict__ gw,
                 const float* __restrict__ gb, float* __restrict__ wmat,
                 float* __restrict__ gpart)
{
    __shared__ float red[256];
    int t = blockIdx.x * 256 + threadIdx.x;

    float lg[8];
    #pragma unroll
    for (int e = 0; e < 8; ++e) lg[e] = gb[e];
    const float* pr = patches + (size_t)t * 768;
    for (int k = 0; k < 768; ++k) {
        float p = pr[k];
        const float* g = gw + k * 8;
        #pragma unroll
        for (int e = 0; e < 8; ++e) lg[e] += p * g[e];
    }
    int i0 = 0; float v0 = lg[0];
    #pragma unroll
    for (int e = 1; e < 8; ++e) if (lg[e] > v0) { v0 = lg[e]; i0 = e; }
    int i1 = (i0 == 0) ? 1 : 0; float v1 = lg[i1];
    #pragma unroll
    for (int e = 0; e < 8; ++e) if (e != i0 && lg[e] > v1) { v1 = lg[e]; i1 = e; }

    float e1 = expf(v1 - v0);
    float den = 1.f + e1;
    wmat[(size_t)i0 * 4096 + t] = 1.f / den;
    wmat[(size_t)i1 * 4096 + t] = e1 / den;

    float pe[8]; float se = 0.f;
    #pragma unroll
    for (int e = 0; e < 8; ++e) { pe[e] = expf(lg[e] - v0); se += pe[e]; }
    float inv = 1.f / se;

    for (int e = 0; e < 8; ++e) {
        red[threadIdx.x] = pe[e] * inv;
        __syncthreads();
        for (int s = 128; s > 0; s >>= 1) {
            if (threadIdx.x < s) red[threadIdx.x] += red[threadIdx.x + s];
            __syncthreads();
        }
        if (threadIdx.x == 0) gpart[blockIdx.x * 16 + e] = red[0];
        __syncthreads();
    }
    for (int e = 0; e < 8; ++e) {
        red[threadIdx.x] = (i0 == e) ? 1.f : 0.f;
        __syncthreads();
        for (int s = 128; s > 0; s >>= 1) {
            if (threadIdx.x < s) red[threadIdx.x] += red[threadIdx.x + s];
            __syncthreads();
        }
        if (threadIdx.x == 0) gpart[blockIdx.x * 16 + 8 + e] = red[0];
        __syncthreads();
    }
}

__global__ void laux_kernel(const float* __restrict__ gpart, float* __restrict__ out)
{
    if (blockIdx.x == 0 && threadIdx.x == 0) {
        float l = 0.f;
        for (int e = 0; e < 8; ++e) {
            float ps = 0.f, cs = 0.f;
            for (int b = 0; b < 16; ++b) { ps += gpart[b * 16 + e]; cs += gpart[b * 16 + 8 + e]; }
            l += (ps / 4096.f) * (cs / 4096.f);
        }
        out[0] = 8.f * l;
    }
}

// ---------------------------------------------------------------------------
// Depthwise 3x3, stride 2, pad 1, ReLU. Output NHWC with channel stride oC.
// ---------------------------------------------------------------------------
__global__ void dw_conv(const float* __restrict__ in, const float* __restrict__ w,
                        const float* __restrict__ b, float* __restrict__ out,
                        int T, int C, int IH, int IW, int OH, int OW, int chw, int oC)
{
    long long idx = (long long)blockIdx.x * blockDim.x + threadIdx.x;
    long long total = (long long)T * OH * OW * C;
    if (idx >= total) return;
    int c = (int)(idx % C);
    long long r = idx / C;
    int ox = (int)(r % OW); r /= OW;
    int oy = (int)(r % OH);
    int t  = (int)(r / OH);
    float acc = b[c];
    #pragma unroll
    for (int ky = 0; ky < 3; ++ky) {
        int iy = oy * 2 - 1 + ky;
        if (iy < 0 || iy >= IH) continue;
        #pragma unroll
        for (int kx = 0; kx < 3; ++kx) {
            int ix = ox * 2 - 1 + kx;
            if (ix < 0 || ix >= IW) continue;
            float v = chw ? in[(((size_t)t * C + c) * IH + iy) * IW + ix]
                          : in[(((size_t)t * IH + iy) * IW + ix) * C + c];
            acc += v * w[c * 9 + ky * 3 + kx];
        }
    }
    out[(((size_t)t * OH + oy) * OW + ox) * oC + c] = fmaxf(acc, 0.f);
}

// 2x2 average pool on NHWC [T,2,2,C] -> feats [T,C]
__global__ void pool2x2(const float* __restrict__ a, float* __restrict__ feats, int T, int C)
{
    long long idx = (long long)blockIdx.x * blockDim.x + threadIdx.x;
    if (idx >= (long long)T * C) return;
    int c = (int)(idx % C); int t = (int)(idx / C);
    size_t b = (size_t)t * 4 * C + c;
    feats[idx] = 0.25f * (a[b] + a[b + C] + a[b + 2 * C] + a[b + 3 * C]);
}

// moe[t,:] += wcol[t] * tmp[t,:]
__global__ void combine_kernel(float* __restrict__ moe, const float* __restrict__ tmp,
                               const float* __restrict__ wcol)
{
    long long idx = (long long)blockIdx.x * blockDim.x + threadIdx.x;
    if (idx >= 4096LL * 768) return;
    int t = (int)(idx / 768);
    moe[idx] += wcol[t] * tmp[idx];
}

// ---------------------------------------------------------------------------
// Skip branch helpers
// ---------------------------------------------------------------------------
__device__ __forceinline__ float skip_conv_px(const float* __restrict__ x,
                                              const float* __restrict__ sw,
                                              const float* __restrict__ sb,
                                              int n, int c, int y, int xx)
{
    float acc = sb[c];
    #pragma unroll
    for (int ky = 0; ky < 3; ++ky) {
        int iy = y + ky - 1;
        if (iy < 0 || iy >= 256) continue;
        #pragma unroll
        for (int kx = 0; kx < 3; ++kx) {
            int ix = xx + kx - 1;
            if (ix < 0 || ix >= 256) continue;
            #pragma unroll
            for (int ci = 0; ci < 3; ++ci)
                acc += x[(((size_t)n * 3 + ci) * 256 + iy) * 256 + ix]
                     * sw[(c * 3 + ci) * 9 + ky * 3 + kx];
        }
    }
    return acc;
}

__global__ __launch_bounds__(256)
void skip_partial(const float* __restrict__ x, const float* __restrict__ sw,
                  const float* __restrict__ sb, float* __restrict__ part)
{
    __shared__ float rs[256], rq[256];
    int c = blockIdx.x;
    int chunk = blockIdx.y;
    float s = 0.f, q = 0.f;
    for (int i = threadIdx.x; i < 16384; i += 256) {
        int p = chunk * 16384 + i;
        int n = p >> 16; int rem = p & 65535; int y = rem >> 8; int xx = rem & 255;
        float v = skip_conv_px(x, sw, sb, n, c, y, xx);
        s += v; q += v * v;
    }
    rs[threadIdx.x] = s; rq[threadIdx.x] = q;
    __syncthreads();
    for (int st = 128; st > 0; st >>= 1) {
        if (threadIdx.x < st) { rs[threadIdx.x] += rs[threadIdx.x + st];
                                rq[threadIdx.x] += rq[threadIdx.x + st]; }
        __syncthreads();
    }
    if (threadIdx.x == 0) {
        part[(c * 64 + chunk) * 2]     = rs[0];
        part[(c * 64 + chunk) * 2 + 1] = rq[0];
    }
}

__global__ __launch_bounds__(64)
void skip_pool(const float* __restrict__ x, const float* __restrict__ sw,
               const float* __restrict__ sb, const float* __restrict__ g,
               const float* __restrict__ be, const float* __restrict__ sum,
               const float* __restrict__ sq, float* __restrict__ h)
{
    int c = threadIdx.x;
    int cell = blockIdx.x;
    int px = cell & 15, py = (cell >> 4) & 15, n = cell >> 8;
    const float invM = 1.f / 1048576.f;
    float mean = sum[c] * invM;
    float var  = sq[c] * invM - mean * mean;
    float inv  = rsqrtf(var + 1e-5f);
    float gg = g[c], bb = be[c];
    float acc = 0.f;
    for (int dy = 0; dy < 16; ++dy)
        for (int dx = 0; dx < 16; ++dx) {
            float v = skip_conv_px(x, sw, sb, n, c, py * 16 + dy, px * 16 + dx);
            v = gg * (v - mean) * inv + bb;
            acc += fmaxf(v, 0.f);
        }
    h[(size_t)cell * 320 + 256 + c] = acc * (1.f / 256.f);
}

// ---------------------------------------------------------------------------
// BN (training mode, biased var): deterministic two-stage per-channel reduce
// ---------------------------------------------------------------------------
__global__ __launch_bounds__(256)
void bn_partial(const float* __restrict__ x, int M, int C,
                float* __restrict__ part, int nch)
{
    __shared__ float rs[256], rq[256];
    int c = blockIdx.x;
    int chunk = blockIdx.y;
    int per = (M + nch - 1) / nch;
    int r0 = chunk * per;
    int r1 = r0 + per; if (r1 > M) r1 = M;
    float s = 0.f, q = 0.f;
    for (int r = r0 + threadIdx.x; r < r1; r += 256) {
        float v = x[(size_t)r * C + c];
        s += v; q += v * v;
    }
    rs[threadIdx.x] = s; rq[threadIdx.x] = q;
    __syncthreads();
    for (int st = 128; st > 0; st >>= 1) {
        if (threadIdx.x < st) { rs[threadIdx.x] += rs[threadIdx.x + st];
                                rq[threadIdx.x] += rq[threadIdx.x + st]; }
        __syncthreads();
    }
    if (threadIdx.x == 0) {
        part[(c * nch + chunk) * 2]     = rs[0];
        part[(c * nch + chunk) * 2 + 1] = rq[0];
    }
}

__global__ void bn_finalize(const float* __restrict__ part, int nch, int C,
                            float* __restrict__ sum, float* __restrict__ sq)
{
    int c = blockIdx.x * blockDim.x + threadIdx.x;
    if (c >= C) return;
    float s = 0.f, q = 0.f;
    for (int k = 0; k < nch; ++k) {
        s += part[(c * nch + k) * 2];
        q += part[(c * nch + k) * 2 + 1];
    }
    sum[c] = s; sq[c] = q;
}

__global__ void bn_apply(float* __restrict__ x, const float* __restrict__ g,
                         const float* __restrict__ be, const float* __restrict__ sum,
                         const float* __restrict__ sq, int M, int C, int relu)
{
    long long idx = (long long)blockIdx.x * blockDim.x + threadIdx.x;
    if (idx >= (long long)M * C) return;
    int c = (int)(idx % C);
    float mean = sum[c] / (float)M;
    float var  = sq[c] / (float)M - mean * mean;
    float v = g[c] * (x[idx] - mean) * rsqrtf(var + 1e-5f) + be[c];
    if (relu) v = fmaxf(v, 0.f);
    x[idx] = v;
}

// ---------------------------------------------------------------------------
// im2col for 3x3 pad1 on a 16x16 NHWC map set ([4096 rows, C]) -> col[4096, C*9]
// ---------------------------------------------------------------------------
__global__ void im2col16(const float* __restrict__ in, float* __restrict__ col, int C)
{
    long long idx = (long long)blockIdx.x * blockDim.x + threadIdx.x;
    long long total = 4096LL * C * 9;
    if (idx >= total) return;
    int Kc = C * 9;
    int k = (int)(idx % Kc);
    int m = (int)(idx / Kc);
    int c = k / 9, r = k % 9, ky = r / 3, kx = r % 3;
    int x = m & 15, y = (m >> 4) & 15, n = m >> 8;
    int iy = y + ky - 1, ix = x + kx - 1;
    float v = 0.f;
    if (iy >= 0 && iy < 16 && ix >= 0 && ix < 16)
        v = in[((size_t)(n * 256 + iy * 16 + ix)) * C + c];
    col[idx] = v;
}

// ---------------------------------------------------------------------------
// ConvTranspose2d k=4 s=2 p=1, NHWC. w: [CI][CO][4][4]
// ---------------------------------------------------------------------------
__global__ void convt4x2(const float* __restrict__ in, const float* __restrict__ w,
                         const float* __restrict__ bias, float* __restrict__ out,
                         int Nn, int IH, int IW, int CI, int CO)
{
    int OH = IH * 2, OW = IW * 2;
    long long idx = (long long)blockIdx.x * blockDim.x + threadIdx.x;
    long long total = (long long)Nn * OH * OW * CO;
    if (idx >= total) return;
    int co = (int)(idx % CO);
    long long r = idx / CO;
    int ox = (int)(r % OW); r /= OW;
    int oy = (int)(r % OH);
    int n  = (int)(r / OH);
    float acc = bias[co];
    #pragma unroll
    for (int ky = 0; ky < 4; ++ky) {
        int ty = oy + 1 - ky;
        if (ty < 0 || (ty & 1)) continue;
        int iy = ty >> 1; if (iy >= IH) continue;
        #pragma unroll
        for (int kx = 0; kx < 4; ++kx) {
            int tx = ox + 1 - kx;
            if (tx < 0 || (tx & 1)) continue;
            int ix = tx >> 1; if (ix >= IW) continue;
            const float* ip = in + ((size_t)(n * IH + iy) * IW + ix) * CI;
            for (int ci = 0; ci < CI; ++ci)
                acc += ip[ci] * w[((size_t)(ci * CO + co) << 4) + ky * 4 + kx];
        }
    }
    out[idx] = acc;
}

// Direct 3x3 pad1 conv, NHWC. w: [CO][CI][3][3]
__global__ void conv3x3_nhwc(const float* __restrict__ in, const float* __restrict__ w,
                             const float* __restrict__ bias, float* __restrict__ out,
                             int Nn, int H, int W, int CI, int CO)
{
    long long idx = (long long)blockIdx.x * blockDim.x + threadIdx.x;
    long long total = (long long)Nn * H * W * CO;
    if (idx >= total) return;
    int co = (int)(idx % CO);
    long long r = idx / CO;
    int x = (int)(r % W); r /= W;
    int y = (int)(r % H);
    int n = (int)(r / H);
    float acc = bias[co];
    #pragma unroll
    for (int ky = 0; ky < 3; ++ky) {
        int iy = y + ky - 1;
        if (iy < 0 || iy >= H) continue;
        #pragma unroll
        for (int kx = 0; kx < 3; ++kx) {
            int ix = x + kx - 1;
            if (ix < 0 || ix >= W) continue;
            const float* ip = in + ((size_t)(n * H + iy) * W + ix) * CI;
            for (int ci = 0; ci < CI; ++ci)
                acc += ip[ci] * w[((size_t)co * CI + ci) * 9 + ky * 3 + kx];
        }
    }
    out[idx] = acc;
}

// fc2: 1x1 conv 8->1 -> d_out image portion
__global__ void fc2_kernel(const float* __restrict__ f1, const float* __restrict__ w,
                           const float* __restrict__ b, float* __restrict__ out)
{
    long long idx = (long long)blockIdx.x * blockDim.x + threadIdx.x;
    if (idx >= 1048576LL) return;
    const float* ip = f1 + idx * 8;
    float acc = b[0];
    #pragma unroll
    for (int c = 0; c < 8; ++c) acc += ip[c] * w[c];
    out[idx] = acc;
}

// ---------------------------------------------------------------------------
// Host driver
// ---------------------------------------------------------------------------
extern "C" void kernel_launch(void* const* d_in, const int* in_sizes, int n_in,
                              void* d_out, int out_size, void* d_ws, size_t ws_size,
                              hipStream_t stream)
{
    (void)in_sizes; (void)n_in; (void)out_size; (void)ws_size;
    const float* x        = (const float*)d_in[0];
    const float* dw1_w    = (const float*)d_in[1];
    const float* dw1_b    = (const float*)d_in[2];
    const float* pw1_w    = (const float*)d_in[3];
    const float* pw1_b    = (const float*)d_in[4];
    const float* dw2_w    = (const float*)d_in[5];
    const float* dw2_b    = (const float*)d_in[6];
    const float* pw2_w    = (const float*)d_in[7];
    const float* pw2_b    = (const float*)d_in[8];
    const float* dw3_w    = (const float*)d_in[9];
    const float* dw3_b    = (const float*)d_in[10];
    const float* pw3_w    = (const float*)d_in[11];
    const float* pw3_b    = (const float*)d_in[12];
    const float* th_w     = (const float*)d_in[13];
    const float* th_b     = (const float*)d_in[14];
    const float* res_w    = (const float*)d_in[15];
    const float* res_b    = (const float*)d_in[16];
    const float* gate_w   = (const float*)d_in[17];
    const float* gate_b   = (const float*)d_in[18];
    const float* to_emb_w = (const float*)d_in[19];
    const float* to_emb_b = (const float*)d_in[20];
    const float* skip_w   = (const float*)d_in[21];
    const float* skip_b   = (const float*)d_in[22];
    const float* skip_g   = (const float*)d_in[23];
    const float* skip_be  = (const float*)d_in[24];
    const float* dp1_w    = (const float*)d_in[25];
    const float* dp1_b    = (const float*)d_in[26];
    const float* dp1_g    = (const float*)d_in[27];
    const float* dp1_be   = (const float*)d_in[28];
    const float* dp2_w    = (const float*)d_in[29];
    const float* dp2_b    = (const float*)d_in[30];
    const float* dp2_g    = (const float*)d_in[31];
    const float* dp2_be   = (const float*)d_in[32];
    const float* up1_w    = (const float*)d_in[33];
    const float* up1_b    = (const float*)d_in[34];
    const float* up1_g    = (const float*)d_in[35];
    const float* up1_be   = (const float*)d_in[36];
    const float* up2_w    = (const float*)d_in[37];
    const float* up2_b    = (const float*)d_in[38];
    const float* up2_g    = (const float*)d_in[39];
    const float* up2_be   = (const float*)d_in[40];
    const float* up3_w    = (const float*)d_in[41];
    const float* up3_b    = (const float*)d_in[42];
    const float* up3_g    = (const float*)d_in[43];
    const float* up3_be   = (const float*)d_in[44];
    const float* up4_w    = (const float*)d_in[45];
    const float* up4_b    = (const float*)d_in[46];
    const float* up4_g    = (const float*)d_in[47];
    const float* up4_be   = (const float*)d_in[48];
    const float* fc1_w    = (const float*)d_in[49];
    const float* fc1_b    = (const float*)d_in[50];
    const float* fc1_g    = (const float*)d_in[51];
    const float* fc1_be   = (const float*)d_in[52];
    const float* fc2_w    = (const float*)d_in[53];
    const float* fc2_b    = (const float*)d_in[54];

    // ---- workspace bump allocator (two overlapping phase arenas) ----
    char* base = (char*)d_ws;
    size_t off = 0;
    auto alloc = [&](size_t nfloat) -> float* {
        float* r = (float*)(base + off);
        off += ((nfloat * sizeof(float)) + 255) & ~(size_t)255;
        return r;
    };
    float* patches = alloc(3145728);   // [4096,768]
    float* wmat    = alloc(32768);     // [8][4096]
    float* gpart   = alloc(256);
    float* moe     = alloc(3145728);   // [4096,768]
    float* h       = alloc(1310720);   // [4096,320]
    float* spart   = alloc(32768);
    float* ssum    = alloc(256);
    float* ssq     = alloc(256);
    float* pw1pad  = alloc(2048);      // padded [64,32] pw1 weights
    size_t arena0 = off;
    // phase 1: expert pipeline
    float* dwo1 = alloc(8388608);      // [4096*64, 32] zero-padded K
    float* a1   = alloc(16777216);     // [4096,8,8,64]
    float* dwo2 = alloc(4194304);      // [4096,4,4,64]
    float* a2   = alloc(8388608);      // [4096,4,4,128]
    float* dwo3 = alloc(2097152);      // [4096,2,2,128]
    float* a3   = alloc(4194304);      // [4096,2,2,256]
    float* feats= alloc(1048576);      // [4096,256]
    float* tmp  = alloc(3145728);      // [4096,768]
    // phase 2: decoder (reuses phase-1 arena)
    off = arena0;
    float* col = alloc(11796480);      // im2col, max [4096,2880]
    float* t_a = alloc(1048576);
    float* t_b = alloc(1048576);
    float* u1  = alloc(2097152);
    float* u2  = alloc(4194304);
    float* u3  = alloc(8388608);
    float* u4  = alloc(16777216);
    float* f1  = alloc(8388608);

    auto gemm = [&](const float* A, const float* Bt, const float* bias, float* C,
                    int M, int N, int K, int lda, int ldb, int ldc, int accum, int relu) {
        int blocks = (M / BM) * ((N + BN - 1) / BN);
        if (accum)
            gemm_wmma_f16<1, 0><<<blocks, 256, 0, stream>>>(A, Bt, bias, C, M, N, K, lda, ldb, ldc);
        else if (relu)
            gemm_wmma_f16<0, 1><<<blocks, 256, 0, stream>>>(A, Bt, bias, C, M, N, K, lda, ldb, ldc);
        else
            gemm_wmma_f16<0, 0><<<blocks, 256, 0, stream>>>(A, Bt, bias, C, M, N, K, lda, ldb, ldc);
    };

    hipMemsetAsync(wmat, 0, 32768 * sizeof(float), stream);
    hipMemsetAsync(moe,  0, 3145728ull * sizeof(float), stream);
    hipMemsetAsync(dwo1, 0, 8388608ull * sizeof(float), stream);

    unfold_kernel<<<ceilDiv(3145728, 256), 256, 0, stream>>>(x, patches);
    gate_kernel<<<16, 256, 0, stream>>>(patches, gate_w, gate_b, wmat, gpart);
    laux_kernel<<<1, 32, 0, stream>>>(gpart, (float*)d_out + 1048576);

    // ---- dense expert evaluation with on-the-fly top-2 weighted combine ----
    for (int e = 0; e < 8; ++e) {
        pad_pw1<<<8, 256, 0, stream>>>(pw1_w + e * 192, pw1pad);
        dw_conv<<<ceilDiv(786432, 256), 256, 0, stream>>>(
            patches, dw1_w + e * 27, dw1_b + e * 3, dwo1, 4096, 3, 16, 16, 8, 8, 1, 32);
        gemm(dwo1, pw1pad, pw1_b + e * 64, a1, 262144, 64, 32, 32, 32, 64, 0, 1);
        dw_conv<<<ceilDiv(4194304, 256), 256, 0, stream>>>(
            a1, dw2_w + e * 576, dw2_b + e * 64, dwo2, 4096, 64, 8, 8, 4, 4, 0, 64);
        gemm(dwo2, pw2_w + e * 8192, pw2_b + e * 128, a2, 65536, 128, 64, 64, 64, 128, 0, 1);
        dw_conv<<<ceilDiv(2097152, 256), 256, 0, stream>>>(
            a2, dw3_w + e * 1152, dw3_b + e * 128, dwo3, 4096, 128, 4, 4, 2, 2, 0, 128);
        gemm(dwo3, pw3_w + e * 32768, pw3_b + e * 256, a3, 16384, 256, 128, 128, 128, 256, 0, 1);
        pool2x2<<<ceilDiv(1048576, 256), 256, 0, stream>>>(a3, feats, 4096, 256);
        gemm(patches, res_w + (size_t)e * 589824, res_b + e * 768, tmp,
             4096, 768, 768, 768, 768, 768, 0, 0);
        gemm(feats, th_w + (size_t)e * 196608, th_b + e * 768, tmp,
             4096, 768, 256, 256, 256, 768, 1, 0);
        combine_kernel<<<ceilDiv(3145728, 256), 256, 0, stream>>>(moe, tmp, wmat + e * 4096);
    }

    // ---- to_emb into concat buffer (cols 0..255 of h, ldc=320) ----
    gemm(moe, to_emb_w, to_emb_b, h, 4096, 256, 768, 768, 768, 320, 0, 0);

    // ---- skip branch into cols 256..319 of h ----
    skip_partial<<<dim3(64, 64), 256, 0, stream>>>(x, skip_w, skip_b, spart);
    bn_finalize<<<1, 64, 0, stream>>>(spart, 64, 64, ssum, ssq);
    skip_pool<<<4096, 64, 0, stream>>>(x, skip_w, skip_b, skip_g, skip_be, ssum, ssq, h);

    // ---- dp1 ----
    im2col16<<<ceilDiv(11796480, 256), 256, 0, stream>>>(h, col, 320);
    gemm(col, dp1_w, dp1_b, t_a, 4096, 256, 2880, 2880, 2880, 256, 0, 0);
    bn_partial<<<dim3(256, 64), 256, 0, stream>>>(t_a, 4096, 256, spart, 64);
    bn_finalize<<<1, 256, 0, stream>>>(spart, 64, 256, ssum, ssq);
    bn_apply<<<ceilDiv(1048576, 256), 256, 0, stream>>>(t_a, dp1_g, dp1_be, ssum, ssq, 4096, 256, 1);

    // ---- dp2 ----
    im2col16<<<ceilDiv(9437184, 256), 256, 0, stream>>>(t_a, col, 256);
    gemm(col, dp2_w, dp2_b, t_b, 4096, 256, 2304, 2304, 2304, 256, 0, 0);
    bn_partial<<<dim3(256, 64), 256, 0, stream>>>(t_b, 4096, 256, spart, 64);
    bn_finalize<<<1, 256, 0, stream>>>(spart, 64, 256, ssum, ssq);
    bn_apply<<<ceilDiv(1048576, 256), 256, 0, stream>>>(t_b, dp2_g, dp2_be, ssum, ssq, 4096, 256, 1);

    // ---- up1..up4 ----
    convt4x2<<<ceilDiv(2097152, 256), 256, 0, stream>>>(t_b, up1_w, up1_b, u1, 16, 16, 16, 256, 128);
    bn_partial<<<dim3(128, 64), 256, 0, stream>>>(u1, 16384, 128, spart, 64);
    bn_finalize<<<1, 128, 0, stream>>>(spart, 64, 128, ssum, ssq);
    bn_apply<<<ceilDiv(2097152, 256), 256, 0, stream>>>(u1, up1_g, up1_be, ssum, ssq, 16384, 128, 1);

    convt4x2<<<ceilDiv(4194304, 256), 256, 0, stream>>>(u1, up2_w, up2_b, u2, 16, 32, 32, 128, 64);
    bn_partial<<<dim3(64, 64), 256, 0, stream>>>(u2, 65536, 64, spart, 64);
    bn_finalize<<<1, 64, 0, stream>>>(spart, 64, 64, ssum, ssq);
    bn_apply<<<ceilDiv(4194304, 256), 256, 0, stream>>>(u2, up2_g, up2_be, ssum, ssq, 65536, 64, 1);

    convt4x2<<<ceilDiv(8388608, 256), 256, 0, stream>>>(u2, up3_w, up3_b, u3, 16, 64, 64, 64, 32);
    bn_partial<<<dim3(32, 64), 256, 0, stream>>>(u3, 262144, 32, spart, 64);
    bn_finalize<<<1, 32, 0, stream>>>(spart, 64, 32, ssum, ssq);
    bn_apply<<<ceilDiv(8388608, 256), 256, 0, stream>>>(u3, up3_g, up3_be, ssum, ssq, 262144, 32, 1);

    convt4x2<<<ceilDiv(16777216, 256), 256, 0, stream>>>(u3, up4_w, up4_b, u4, 16, 128, 128, 32, 16);
    bn_partial<<<dim3(16, 64), 256, 0, stream>>>(u4, 1048576, 16, spart, 64);
    bn_finalize<<<1, 16, 0, stream>>>(spart, 64, 16, ssum, ssq);
    bn_apply<<<ceilDiv(16777216, 256), 256, 0, stream>>>(u4, up4_g, up4_be, ssum, ssq, 1048576, 16, 1);

    // ---- fc1 + BN/ReLU ----
    conv3x3_nhwc<<<ceilDiv(8388608, 256), 256, 0, stream>>>(u4, fc1_w, fc1_b, f1, 16, 256, 256, 16, 8);
    bn_partial<<<dim3(8, 64), 256, 0, stream>>>(f1, 1048576, 8, spart, 64);
    bn_finalize<<<1, 8, 0, stream>>>(spart, 64, 8, ssum, ssq);
    bn_apply<<<ceilDiv(8388608, 256), 256, 0, stream>>>(f1, fc1_g, fc1_be, ssum, ssq, 1048576, 8, 1);

    // ---- fc2 -> output image ----
    fc2_kernel<<<ceilDiv(1048576, 256), 256, 0, stream>>>(f1, fc2_w, fc2_b, (float*)d_out);
}
